// MambaMixer_19301583029039
// MI455X (gfx1250) — matrix operand order repaired and checked by reading, not compiled, not run on hardware
//
#include <hip/hip_runtime.h>
#include <hip/hip_bf16.h>

// ---------------------------------------------------------------------------
// Mamba-2 mixer block for MI455X (gfx1250, wave32).
// All matmul-shaped contractions run on v_wmma_f32_16x16x32_bf16 (f32 accum).
// LDS layouts give contiguous per-lane fragments (ds_load_b128 x2):
//   A operand: row-major [M][K];  B operand: column-major [N][K].
// Staging uses vectorized ds_store_b128/b64/b32 where global coalescing allows.
// ---------------------------------------------------------------------------

typedef __bf16 bf16x16 __attribute__((ext_vector_type(16)));
typedef __bf16 bf16x8 __attribute__((ext_vector_type(8)));
typedef __bf16 bf16x4 __attribute__((ext_vector_type(4)));
typedef __bf16 bf16x2 __attribute__((ext_vector_type(2)));
typedef float f32x8 __attribute__((ext_vector_type(8)));

#define BLn 4096        // B*L rows
#define DMODEL 2048
#define DINNER 4096
#define NHEADS 64
#define HEADDIM 64
#define DSTATE 128
#define CONVDIM 4352
#define DINPROJ 8512
#define CHUNK 256
#define NCHUNK 8

__device__ __forceinline__ f32x8 wmma_bf16(bf16x16 a, bf16x16 b, f32x8 c) {
  // 8-arg form: (neg_a, A, neg_b, B, c_mod, C, reuse_a, reuse_b)
  return __builtin_amdgcn_wmma_f32_16x16x32_bf16(false, a, false, b, (short)0, c,
                                                 false, false);
}

// A fragment: 16x32 bf16 tile from row-major LDS (row stride ld).
__device__ __forceinline__ bf16x16 load_a_frag(const __bf16* lds, int row0, int k0,
                                               int ld) {
  int lane = threadIdx.x & 31;
  int m = lane & 15, lh = lane >> 4;
  const __bf16* base = lds + (size_t)(row0 + m) * ld + k0 + lh * 16;
  bf16x8 lo = *(const bf16x8*)(base);
  bf16x8 hi = *(const bf16x8*)(base + 8);
  return __builtin_shufflevector(lo, hi, 0, 1, 2, 3, 4, 5, 6, 7, 8, 9, 10, 11, 12,
                                 13, 14, 15);
}

// B fragment: 32x16 (KxN) tile from COLUMN-MAJOR LDS [N][ldk]; 32 contiguous
// bytes per lane -> ds_load_b128 x2.
__device__ __forceinline__ bf16x16 load_b_frag_cm(const __bf16* lds, int k0, int col0,
                                                  int ldk) {
  int lane = threadIdx.x & 31;
  int n = lane & 15, kb = (lane >> 4) * 16;
  const __bf16* base = lds + (size_t)(col0 + n) * ldk + k0 + kb;
  bf16x8 lo = *(const bf16x8*)(base);
  bf16x8 hi = *(const bf16x8*)(base + 8);
  return __builtin_shufflevector(lo, hi, 0, 1, 2, 3, 4, 5, 6, 7, 8, 9, 10, 11, 12,
                                 13, 14, 15);
}

// ---------------------------------------------------------------------------
// Generic bf16-WMMA GEMM: C[M,N] = A[M,K] * B[K,N], f32 output.
// Block tile 256x64, BK=32, 256 threads = 8 waves.
// Each wave: 32 rows x 64 cols = 2 A-frags x 4 B-frags = 8 WMMA per K-step.
// ---------------------------------------------------------------------------
template <typename TA, typename TB>
__global__ void gemm_bf16_wmma(const TA* __restrict__ A, const TB* __restrict__ B,
                               float* __restrict__ C, int M, int N, int K) {
  __shared__ __bf16 As[256 * 32];  // row-major [m][k]
  __shared__ __bf16 Bs[64 * 32];   // column-major [n][k]
  const int t = threadIdx.x;
  const int wave = t >> 5, lane = t & 31;
  const int lh = lane >> 4, nn = lane & 15;
  const int m0 = blockIdx.x * 256, n0 = blockIdx.y * 64;
  f32x8 acc[2][4] = {};
  for (int k0 = 0; k0 < K; k0 += 32) {
    // A tile: each thread converts 8 consecutive floats -> one ds_store_b128
#pragma unroll
    for (int r = 0; r < 4; ++r) {
      int chunk = t + r * 256;                    // 1024 chunks of 8
      int row = chunk >> 2, coff = (chunk & 3) * 8;
      const TA* g = &A[(size_t)(m0 + row) * K + k0 + coff];
      bf16x8 v;
#pragma unroll
      for (int i = 0; i < 8; ++i) v[i] = (__bf16)(float)g[i];
      *(bf16x8*)&As[row * 32 + coff] = v;
    }
    // B tile: each thread owns (col, k-quad) -> one ds_store_b64; global reads
    // stay coalesced (consecutive threads -> consecutive cols).
#pragma unroll
    for (int r = 0; r < 2; ++r) {
      int item = t + r * 256;                     // 512 items
      int col = item & 63, kq = item >> 6;
      bf16x4 v;
#pragma unroll
      for (int i = 0; i < 4; ++i)
        v[i] = (__bf16)(float)B[(size_t)(k0 + kq * 4 + i) * N + n0 + col];
      *(bf16x4*)&Bs[col * 32 + kq * 4] = v;
    }
    if (k0 + 32 < K) {  // speculative prefetch of next K-slab -> global_prefetch_b8
      __builtin_prefetch(&A[(size_t)(m0 + (t & 255)) * K + k0 + 32], 0, 1);
      __builtin_prefetch(&B[(size_t)(k0 + 32 + (t & 31)) * N + n0 + (t >> 5)], 0, 1);
    }
    __syncthreads();
    bf16x16 a0 = load_a_frag(As, wave * 32, 0, 32);
    bf16x16 a1 = load_a_frag(As, wave * 32 + 16, 0, 32);
    bf16x16 bfr[4];
#pragma unroll
    for (int j = 0; j < 4; ++j) bfr[j] = load_b_frag_cm(Bs, 0, j * 16, 32);
#pragma unroll
    for (int j = 0; j < 4; ++j) {
      acc[0][j] = wmma_bf16(a0, bfr[j], acc[0][j]);
      acc[1][j] = wmma_bf16(a1, bfr[j], acc[1][j]);
    }
    __syncthreads();
  }
#pragma unroll
  for (int i = 0; i < 2; ++i)
#pragma unroll
    for (int j = 0; j < 4; ++j)
#pragma unroll
      for (int e = 0; e < 8; ++e)
        C[(size_t)(m0 + wave * 32 + i * 16 + e + 8 * lh) * N + n0 + j * 16 + nn] =
            acc[i][j][e];
}

// ---------------------------------------------------------------------------
// Causal conv1d (K=4) + SiLU over the xBC slice of zxbcdt.
// ---------------------------------------------------------------------------
__global__ void conv_silu_kernel(const float* __restrict__ zx,
                                 const float* __restrict__ w,
                                 const float* __restrict__ bias,
                                 float* __restrict__ xbc) {
  size_t idx = (size_t)blockIdx.x * 256 + threadIdx.x;
  if (idx >= (size_t)BLn * CONVDIM) return;
  int ch = (int)(idx % CONVDIM);
  int bl = (int)(idx / CONVDIM);
  int b = bl >> 11, l = bl & 2047;
  float acc = bias[ch];
#pragma unroll
  for (int k = 0; k < 4; ++k) {
    int li = l - 3 + k;
    if (li >= 0)
      acc += zx[(size_t)(b * 2048 + li) * DINPROJ + DINNER + ch] * w[ch * 4 + k];
  }
  xbc[idx] = acc / (1.0f + __expf(-acc));
}

// dt = softplus(dt_raw + dt_bias), layout [bl][h]
__global__ void dt_softplus_kernel(const float* __restrict__ zx,
                                   const float* __restrict__ dt_bias,
                                   float* __restrict__ dt_sp) {
  int idx = blockIdx.x * 256 + threadIdx.x;
  if (idx >= BLn * NHEADS) return;
  int h = idx & 63, bl = idx >> 6;
  float v = zx[(size_t)bl * DINPROJ + (DINNER + CONVDIM) + h] + dt_bias[h];
  dt_sp[idx] = (v > 20.0f) ? v : log1pf(__expf(v));
}

// Inclusive scan of dA = dt*A over each chunk. One block per (b,h,c).
// Acum layout: [b][h][c][l]
__global__ void cumsum_kernel(const float* __restrict__ dt_sp,
                              const float* __restrict__ A,
                              float* __restrict__ Acum) {
  __shared__ float buf[2][CHUNK];
  int bid = blockIdx.x;  // (b*64+h)*8+c
  int c = bid & 7, h = (bid >> 3) & 63, b = bid >> 9;
  int l = threadIdx.x;
  int bl = b * 2048 + c * CHUNK + l;
  buf[0][l] = dt_sp[(size_t)bl * NHEADS + h] * A[h];
  __syncthreads();
  int src = 0;
  for (int off = 1; off < CHUNK; off <<= 1) {
    float x = buf[src][l];
    if (l >= off) x += buf[src][l - off];
    buf[src ^ 1][l] = x;
    src ^= 1;
    __syncthreads();
  }
  Acum[(size_t)bid * CHUNK + l] = buf[src][l];
}

// ---------------------------------------------------------------------------
// chunk_state: S[p,n] = sum_l x[l,p]*dt[l]*exp(Acum[255]-Acum[l]) * B[l,n]
// WMMA GEMM 64x128x256 per (b,c,h). S layout: [(b*8+c)*64+h][p][n]
// ---------------------------------------------------------------------------
__global__ void chunk_state_kernel(const float* __restrict__ xbc,
                                   const float* __restrict__ dt_sp,
                                   const float* __restrict__ Acum,
                                   float* __restrict__ S) {
  __shared__ __bf16 At[64 * 32];   // A = xs^T : row-major [p][l]
  __shared__ __bf16 Bt[128 * 32];  // column-major [n][l]
  __shared__ float coef[CHUNK];
  int bid = blockIdx.x;  // (b*8+c)*64+h
  int h = bid & 63, c = (bid >> 6) & 7, b = bid >> 9;
  int t = threadIdx.x, wave = t >> 5, lane = t & 31;
  int lh = lane >> 4, nn = lane & 15;
  const float* ac = Acum + ((size_t)((b * 64 + h) * 8 + c)) * CHUNK;
  size_t bl0 = (size_t)(b * 2048 + c * CHUNK);
  {
    int l = t;
    float last = ac[CHUNK - 1];
    coef[l] = dt_sp[(bl0 + l) * NHEADS + h] * __expf(last - ac[l]);
  }
  __syncthreads();
  f32x8 acc[4] = {};
  int mt = wave & 3;          // p-tile
  int n0 = (wave >> 2) * 64;  // n offset
  for (int l0 = 0; l0 < CHUNK; l0 += 32) {
    // x tile: (p, l-pair) per thread -> ds_store_b32, global reads coalesced in p
#pragma unroll
    for (int r = 0; r < 4; ++r) {
      int item = t + r * 256;  // 1024 items
      int p = item & 63, lq = item >> 6;
      bf16x2 v;
#pragma unroll
      for (int i = 0; i < 2; ++i) {
        int l = lq * 2 + i;
        v[i] = (__bf16)(xbc[(bl0 + l0 + l) * CONVDIM + h * HEADDIM + p] *
                        coef[l0 + l]);
      }
      *(bf16x2*)&At[p * 32 + lq * 2] = v;
    }
    // B tile: (n, l-quad) per thread -> ds_store_b64, global reads coalesced in n
#pragma unroll
    for (int r = 0; r < 4; ++r) {
      int item = t + r * 256;  // 1024 items
      int n = item & 127, lq = item >> 7;
      bf16x4 v;
#pragma unroll
      for (int i = 0; i < 4; ++i)
        v[i] = (__bf16)xbc[(bl0 + l0 + lq * 4 + i) * CONVDIM + DINNER + n];
      *(bf16x4*)&Bt[n * 32 + lq * 4] = v;
    }
    __syncthreads();
    bf16x16 af = load_a_frag(At, mt * 16, 0, 32);
    bf16x16 bfr[4];
#pragma unroll
    for (int j = 0; j < 4; ++j) bfr[j] = load_b_frag_cm(Bt, 0, n0 + j * 16, 32);
#pragma unroll
    for (int j = 0; j < 4; ++j) acc[j] = wmma_bf16(af, bfr[j], acc[j]);
    __syncthreads();
  }
  float* Sout = S + (size_t)bid * HEADDIM * DSTATE;
#pragma unroll
  for (int j = 0; j < 4; ++j)
#pragma unroll
    for (int e = 0; e < 8; ++e)
      Sout[(mt * 16 + e + 8 * lh) * DSTATE + n0 + j * 16 + nn] = acc[j][e];
}

// Sequential 8-chunk state recurrence: SP_c = exp(dAsum_{c-1})*SP_{c-1} + S_{c-1}
__global__ void state_passing_kernel(const float* __restrict__ S,
                                     const float* __restrict__ Acum,
                                     float* __restrict__ SP) {
  int idx = blockIdx.x * 256 + threadIdx.x;  // b*h*p*n = 1048576
  if (idx >= 2 * 64 * 64 * 128) return;
  int n = idx & 127, p = (idx >> 7) & 63, h = (idx >> 13) & 63, b = idx >> 19;
  float prev = 0.0f;
  for (int c = 0; c < NCHUNK; ++c) {
    size_t off = ((size_t)((b * 8 + c) * 64 + h)) * (HEADDIM * DSTATE) + p * DSTATE + n;
    SP[off] = prev;
    float dAsum = Acum[((size_t)((b * 64 + h) * 8 + c)) * CHUNK + (CHUNK - 1)];
    prev = prev * __expf(dAsum) + S[off];
  }
}

// ---------------------------------------------------------------------------
// chunk_output: Y = (mask(C·B^T)∘L∘dt)·X + exp(Acum)∘(C·SP^T) + D*x
// One block per (b,c,h), 256 threads, ~211 KB dynamic LDS (WGP has 320 KB).
// B-operand buffers are column-major in the WMMA sense, which is simply the
// natural layout of B[s][n] and SP[p][n] (no transpose needed).
// ---------------------------------------------------------------------------
__global__ void chunk_output_kernel(const float* __restrict__ xbc,
                                    const float* __restrict__ dt_sp,
                                    const float* __restrict__ Acum,
                                    const float* __restrict__ SP,
                                    const float* __restrict__ Dskip,
                                    float* __restrict__ Y) {
  extern __shared__ char smem[];
  __bf16* Cs = (__bf16*)smem;            // 256x128  A-operand, row-major [s][n]
  __bf16* Bcm = Cs + 256 * 128;          // 256x128  B-operand cm: [s][n] (raw B!)
  __bf16* XT = Bcm + 256 * 128;          // 64x256   B-operand cm: [p][s]
  __bf16* Gs = XT + 64 * 256;            // 64x256   A-operand, row-major [l][s]
  __bf16* SPcm = Gs + 64 * 256;          // 64x128   B-operand cm: [p][n] (raw SP!)
  float* acum = (float*)(SPcm + 64 * 128);  // 256
  float* dts = acum + 256;                  // 256
  float* eac = dts + 256;                   // 256
  int bid = blockIdx.x;  // (b*8+c)*64+h
  int h = bid & 63, c = (bid >> 6) & 7, b = bid >> 9;
  int t = threadIdx.x, wave = t >> 5, lane = t & 31;
  int lh = lane >> 4, nn = lane & 15;
  size_t bl0 = (size_t)(b * 2048 + c * CHUNK);
  const float* ac = Acum + ((size_t)((b * 64 + h) * 8 + c)) * CHUNK;
  {
    int l = t;
    float a = ac[l];
    acum[l] = a;
    eac[l] = __expf(a);
    dts[l] = dt_sp[(bl0 + l) * NHEADS + h];
  }
  __syncthreads();
  // C + B staging, n-pairs -> ds_store_b32, global reads coalesced
  for (int r = 0; r < 64; ++r) {
    int item = t + r * 256;  // 16384 pairs
    int s = item >> 6, n2 = (item & 63) * 2;
    const float* gC = &xbc[(bl0 + s) * CONVDIM + (DINNER + DSTATE) + n2];
    const float* gB = &xbc[(bl0 + s) * CONVDIM + DINNER + n2];
    bf16x2 vc, vb;
    vc[0] = (__bf16)gC[0]; vc[1] = (__bf16)gC[1];
    vb[0] = (__bf16)gB[0]; vb[1] = (__bf16)gB[1];
    *(bf16x2*)&Cs[s * 128 + n2] = vc;
    *(bf16x2*)&Bcm[s * 128 + n2] = vb;
  }
  // X staging, transpose-on-write -> [p][s] (scalar stores keep global reads
  // coalesced along p; packing along s would de-coalesce HBM reads)
  for (int r = 0; r < 64; ++r) {
    int idx = t + r * 256;
    int s = idx >> 6, p = idx & 63;
    XT[p * 256 + s] = (__bf16)xbc[(bl0 + s) * CONVDIM + h * HEADDIM + p];
  }
  // SP staging, direct copy [p][n], n-pairs -> ds_store_b32
  const float* sp = SP + (size_t)bid * (HEADDIM * DSTATE);
  for (int r = 0; r < 16; ++r) {
    int item = t + r * 256;  // 4096 pairs
    int e2 = item * 2;
    bf16x2 v;
    v[0] = (__bf16)sp[e2]; v[1] = (__bf16)sp[e2 + 1];
    *(bf16x2*)&SPcm[e2] = v;
  }
  __syncthreads();

  int mt = wave & 3;
  float dsk = Dskip[h];
  for (int lt = 0; lt < 4; ++lt) {  // 64-row l-tiles
    // ---- stage 1: G = C_tile(64x128) @ B^T(128x256) ----
    int gn0 = (wave >> 2) * 128;
    f32x8 gt[8] = {};
    for (int k0 = 0; k0 < DSTATE; k0 += 32) {
      bf16x16 af = load_a_frag(Cs, lt * 64 + mt * 16, k0, 128);
      bf16x16 bfr[8];
#pragma unroll
      for (int j = 0; j < 8; ++j) bfr[j] = load_b_frag_cm(Bcm, k0, gn0 + j * 16, 128);
#pragma unroll
      for (int j = 0; j < 8; ++j) gt[j] = wmma_bf16(af, bfr[j], gt[j]);
    }
    // causal decay mask + dt scaling, restage as bf16
#pragma unroll
    for (int j = 0; j < 8; ++j) {
      int s = gn0 + j * 16 + nn;
#pragma unroll
      for (int e = 0; e < 8; ++e) {
        int lrow = mt * 16 + e + 8 * lh;
        int l = lt * 64 + lrow;
        float v = gt[j][e];
        v = (s <= l) ? v * __expf(acum[l] - acum[s]) * dts[s] : 0.0f;
        Gs[lrow * 256 + s] = (__bf16)v;
      }
    }
    __syncthreads();
    // ---- stage 2: Y = G(64x256)@X(256x64) + eac ∘ (C_tile@SP^T) ----
    int yn0 = (wave >> 2) * 32;
    f32x8 ya[2] = {}, yb[2] = {};
    for (int k0 = 0; k0 < CHUNK; k0 += 32) {
      bf16x16 af = load_a_frag(Gs, mt * 16, k0, 256);
#pragma unroll
      for (int j = 0; j < 2; ++j) {
        bf16x16 bf = load_b_frag_cm(XT, k0, yn0 + j * 16, 256);
        ya[j] = wmma_bf16(af, bf, ya[j]);
      }
    }
    for (int k0 = 0; k0 < DSTATE; k0 += 32) {
      bf16x16 af = load_a_frag(Cs, lt * 64 + mt * 16, k0, 128);
#pragma unroll
      for (int j = 0; j < 2; ++j) {
        bf16x16 bf = load_b_frag_cm(SPcm, k0, yn0 + j * 16, 128);
        yb[j] = wmma_bf16(af, bf, yb[j]);
      }
    }
#pragma unroll
    for (int j = 0; j < 2; ++j) {
      int p = yn0 + j * 16 + nn;
#pragma unroll
      for (int e = 0; e < 8; ++e) {
        int lrow = mt * 16 + e + 8 * lh;
        int l = lt * 64 + lrow;
        float xv = xbc[(bl0 + l) * CONVDIM + h * HEADDIM + p];
        Y[(bl0 + l) * DINNER + h * HEADDIM + p] = ya[j][e] + eac[l] * yb[j][e] + dsk * xv;
      }
    }
    __syncthreads();  // Gs reused next lt
  }
}

// ---------------------------------------------------------------------------
// Gated RMSNorm -> bf16 activations for the out-projection GEMM.
// ---------------------------------------------------------------------------
__global__ void rmsnorm_kernel(const float* __restrict__ Y,
                               const float* __restrict__ zx,
                               const float* __restrict__ nw,
                               __bf16* __restrict__ out_bf) {
  __shared__ float wsum[8];
  int row = blockIdx.x, t = threadIdx.x;
  float xv[16];
  float ss = 0.0f;
#pragma unroll
  for (int r = 0; r < 16; ++r) {
    int i = t + r * 256;
    float y = Y[(size_t)row * DINNER + i];
    float z = zx[(size_t)row * DINPROJ + i];
    float v = y * (z / (1.0f + __expf(-z)));
    xv[r] = v;
    ss += v * v;
  }
#pragma unroll
  for (int off = 16; off > 0; off >>= 1) ss += __shfl_xor(ss, off, 32);
  if ((t & 31) == 0) wsum[t >> 5] = ss;
  __syncthreads();
  float tot = 0.0f;
#pragma unroll
  for (int w = 0; w < 8; ++w) tot += wsum[w];
  float scale = rsqrtf(tot / (float)DINNER + 1e-5f);
#pragma unroll
  for (int r = 0; r < 16; ++r) {
    int i = t + r * 256;
    out_bf[(size_t)row * DINNER + i] = (__bf16)(xv[r] * scale * nw[i]);
  }
}

// ---------------------------------------------------------------------------
extern "C" void kernel_launch(void* const* d_in, const int* in_sizes, int n_in,
                              void* d_out, int out_size, void* d_ws, size_t ws_size,
                              hipStream_t stream) {
  (void)in_sizes; (void)n_in; (void)out_size; (void)ws_size;
  const float* H = (const float*)d_in[0];        // (2,2048,2048)
  const float* W_in = (const float*)d_in[1];     // (2048,8512)
  const float* conv_w = (const float*)d_in[2];   // (4352,4)
  const float* conv_b = (const float*)d_in[3];   // (4352,)
  const float* Avec = (const float*)d_in[4];     // (64,)
  const float* Dvec = (const float*)d_in[5];     // (64,)
  const float* dt_bias = (const float*)d_in[6];  // (64,)
  const float* norm_w = (const float*)d_in[7];   // (4096,)
  const float* W_out = (const float*)d_in[8];    // (4096,2048)
  float* out = (float*)d_out;                    // (2,2048,2048) f32
  float* ws = (float*)d_ws;

  size_t off = 0;
  float* zx = ws + off;      off += (size_t)BLn * DINPROJ;           // 34.9M
  float* xbc = ws + off;     off += (size_t)BLn * CONVDIM;           // 17.8M
  float* dt_sp = ws + off;   off += (size_t)BLn * NHEADS;            // 262K
  float* Acum = ws + off;    off += (size_t)2 * 64 * 8 * CHUNK;      // 262K
  float* S = ws + off;       off += (size_t)2 * 8 * 64 * 64 * 128;   // 8.4M
  float* SP = ws + off;      off += (size_t)2 * 8 * 64 * 64 * 128;   // 8.4M
  float* Yb = ws + off;      off += (size_t)BLn * DINNER;            // 16.8M
  __bf16* yn = (__bf16*)(ws + off);                                  // 16.8M bf16

  // 1) in-projection: zx = H @ W_in   (4096 x 8512 x 2048)
  gemm_bf16_wmma<float, float>
      <<<dim3(BLn / 256, DINPROJ / 64), 256, 0, stream>>>(H, W_in, zx, BLn, DINPROJ,
                                                          DMODEL);
  // 2) causal conv1d + SiLU
  conv_silu_kernel<<<((size_t)BLn * CONVDIM) / 256, 256, 0, stream>>>(zx, conv_w,
                                                                      conv_b, xbc);
  // 3) dt softplus
  dt_softplus_kernel<<<(BLn * NHEADS) / 256, 256, 0, stream>>>(zx, dt_bias, dt_sp);
  // 4) per-chunk cumsum of dA
  cumsum_kernel<<<2 * 64 * 8, CHUNK, 0, stream>>>(dt_sp, Avec, Acum);
  // 5) per-chunk states (WMMA)
  chunk_state_kernel<<<2 * 8 * 64, 256, 0, stream>>>(xbc, dt_sp, Acum, S);
  // 6) inter-chunk state recurrence
  state_passing_kernel<<<(2 * 64 * 64 * 128) / 256, 256, 0, stream>>>(S, Acum, SP);
  // 7) chunk outputs (WMMA x3), ~211KB dynamic LDS per workgroup (WGP has 320KB)
  size_t smem = (size_t)(256 * 128 + 256 * 128 + 64 * 256 + 64 * 256 + 64 * 128) *
                    sizeof(__bf16) +
                3 * 256 * sizeof(float);
  chunk_output_kernel<<<2 * 8 * 64, 256, smem, stream>>>(xbc, dt_sp, Acum, SP, Dvec,
                                                         Yb);
  // 8) gated RMSNorm -> bf16
  rmsnorm_kernel<<<BLn, 256, 0, stream>>>(Yb, zx, norm_w, yn);
  // 9) out-projection: out = yn @ W_out  (4096 x 2048 x 4096)
  gemm_bf16_wmma<__bf16, float>
      <<<dim3(BLn / 256, DMODEL / 64), 256, 0, stream>>>(yn, W_out, out, BLn, DMODEL,
                                                         DINNER);
}
